// SelfAttention_13005160972417
// MI455X (gfx1250) — matrix operand order repaired and checked by reading, not compiled, not run on hardware
//
#include <hip/hip_runtime.h>
#include <hip/hip_bf16.h>
#include <math.h>

// MI455X / gfx1250, wave32: WMMA f16 + Tensor Data Mover staging.

typedef __attribute__((ext_vector_type(16))) _Float16 v16h;
typedef __attribute__((ext_vector_type(8)))  _Float16 v8h;
typedef __attribute__((ext_vector_type(8)))  float    v8f;
typedef __attribute__((ext_vector_type(4)))  unsigned int u32x4;
typedef __attribute__((ext_vector_type(8)))  unsigned int u32x8;

#define B_   4
#define L_   2048
#define DIM_ 1024
#define H_   16
#define DH_  64

static __device__ __forceinline__ v16h mkv16(v8h lo, v8h hi) {
  v16h r;
#pragma unroll
  for (int i = 0; i < 8; ++i) { r[i] = lo[i]; r[i + 8] = hi[i]; }
  return r;
}

static __device__ __forceinline__ v8f wmma16(v16h a, v16h b, v8f c) {
  // D = A(16x32 f16) * B(32x16 f16) + C(16x16 f32)
  return __builtin_amdgcn_wmma_f32_16x16x32_f16(false, a, false, b,
                                                (short)0, c, false, false);
}

// Issue one TDM 2-D tile load (global -> LDS), D# per ISA 08_async_tensor §8.
// All-dims in data_size units (2-byte elements); global_addr/lds_addr in bytes.
static __device__ __forceinline__ void tdm_load_2d(
    unsigned lds_addr, unsigned long long gaddr,
    unsigned tensor_d0, unsigned tensor_d1,
    unsigned tile_d0, unsigned tile_d1,
    unsigned long long stride0)
{
  u32x4 g0;
  g0[0] = 1u;                                          // count=1, user D#
  g0[1] = lds_addr;                                    // [63:32] lds_addr
  g0[2] = (unsigned)(gaddr & 0xFFFFFFFFu);             // [95:64] ga lo
  g0[3] = (unsigned)((gaddr >> 32) & 0x1FFFFFFu) | (2u << 30);  // ga hi | type=2
  u32x8 g1;
  g1[0] = (1u << 16);                                  // data_size=1 (2B), mask=0
  g1[1] = (tensor_d0 & 0xFFFFu) << 16;                 // tensor_dim0[15:0]
  g1[2] = ((tensor_d0 >> 16) & 0xFFFFu) | ((tensor_d1 & 0xFFFFu) << 16);
  g1[3] = ((tensor_d1 >> 16) & 0xFFFFu) | ((tile_d0 & 0xFFFFu) << 16);
  g1[4] = (tile_d1 & 0xFFFFu);                         // tile_dim2 = 0
  g1[5] = (unsigned)(stride0 & 0xFFFFFFFFu);           // tensor_dim0_stride lo
  g1[6] = (unsigned)((stride0 >> 32) & 0xFFFFu);       // stride hi; dim1_stride=0
  g1[7] = 0u;
  asm volatile("tensor_load_to_lds %0, %1" :: "s"(g0), "s"(g1) : "memory");
}

// ---------------------------------------------------------------------------
// Kernel 1: fused QKV projection.  C[M=8192, N=3072] = x @ [Wq|Wk|Wv]^T + bias
// 64x128 tile per block: 8 waves, each wave 4 subtiles sharing one A-fragment.
// Writes f16: Q (scaled 0.125) [b,h,l,dh], K [b,h,l,dh], V^T [b,h,dh,l].
// ---------------------------------------------------------------------------
__global__ __launch_bounds__(256) void qkv_wmma_kernel(
    const float* __restrict__ x,
    const float* __restrict__ wq, const float* __restrict__ bq,
    const float* __restrict__ wk, const float* __restrict__ bk,
    const float* __restrict__ wv, const float* __restrict__ bv,
    _Float16* __restrict__ qws, _Float16* __restrict__ kws,
    _Float16* __restrict__ vtws)
{
  __shared__ _Float16 xs[64 * 32];    // A tile [m][k] f16
  __shared__ _Float16 wsh[128 * 32];  // B tile [n][k] f16 (W row-major = B(k,n))

  const int tid  = threadIdx.x;
  const int lane = tid & 31;
  const int wave = tid >> 5;

  const int mBase = blockIdx.y * 64;       // row in [0, 8192)
  const int nBase = blockIdx.x * 128;      // col in [0, 3072)
  const int sel   = nBase >> 10;           // 0=Q, 1=K, 2=V (128 | 1024)
  const int nCol  = nBase & 1023;

  const float* W  = (sel == 0) ? wq : (sel == 1) ? wk : wv;
  const float* Bp = (sel == 0) ? bq : (sel == 1) ? bk : bv;

  const int mi  = wave & 3;                // subtile row 0..3
  const int ni0 = wave >> 2;               // subtile cols {ni0 + 2t}, t=0..3

  v8f acc[4] = {{}, {}, {}, {}};

  // staging: x tile 2048 halves, W tile 4096 halves, 256 threads
  const int srow = tid >> 2;               // 0..63
  const int scol = (tid & 3) * 8;          // 0,8,16,24
  const int wrow = tid >> 1;               // 0..127
  const int wcol = (tid & 1) * 16;         // 0,16

  // fragment addressing (ISA 7.12.2, 16-bit 16x32 A / 32x16 B)
  const int am   = mi * 16 + (lane & 15);  // A row
  const int khi  = (lane >> 4) * 8;        // A k-halves base
  const int bk16 = (lane >> 4) * 8;        // B k base
  const int bnl  = lane & 15;              // B column within subtile

  for (int kb = 0; kb < DIM_; kb += 32) {
    {
      const float4* px = (const float4*)(x + (size_t)(mBase + srow) * DIM_ + kb + scol);
      float4 f0 = px[0], f1 = px[1];
      v8h hx = { (_Float16)f0.x, (_Float16)f0.y, (_Float16)f0.z, (_Float16)f0.w,
                 (_Float16)f1.x, (_Float16)f1.y, (_Float16)f1.z, (_Float16)f1.w };
      *(v8h*)&xs[srow * 32 + scol] = hx;

      const float4* pw = (const float4*)(W + (size_t)(nCol + wrow) * DIM_ + kb + wcol);
      float4 g0 = pw[0], g1 = pw[1], g2 = pw[2], g3 = pw[3];
      v8h hw0 = { (_Float16)g0.x, (_Float16)g0.y, (_Float16)g0.z, (_Float16)g0.w,
                  (_Float16)g1.x, (_Float16)g1.y, (_Float16)g1.z, (_Float16)g1.w };
      v8h hw1 = { (_Float16)g2.x, (_Float16)g2.y, (_Float16)g2.z, (_Float16)g2.w,
                  (_Float16)g3.x, (_Float16)g3.y, (_Float16)g3.z, (_Float16)g3.w };
      *(v8h*)&wsh[wrow * 32 + wcol]     = hw0;
      *(v8h*)&wsh[wrow * 32 + wcol + 8] = hw1;
    }
    __syncthreads();

    v16h A = mkv16(*(const v8h*)&xs[am * 32 + khi],
                   *(const v8h*)&xs[am * 32 + 16 + khi]);
#pragma unroll
    for (int t = 0; t < 4; ++t) {
      const int bn = (ni0 + 2 * t) * 16 + bnl;
      v16h Bf = mkv16(*(const v8h*)&wsh[bn * 32 + bk16],
                      *(const v8h*)&wsh[bn * 32 + bk16 + 16]);
      acc[t] = wmma16(A, Bf, acc[t]);
    }
    __syncthreads();
  }

  // epilogue: bias, Q-scale, head-split scatter stores
  const int rHi = (lane >> 4) * 8;
#pragma unroll
  for (int t = 0; t < 4; ++t) {
    const int n    = nCol + (ni0 + 2 * t) * 16 + bnl;
    const float bias = Bp[n];
    const int hh = n >> 6, dh = n & 63;
#pragma unroll
    for (int r = 0; r < 8; ++r) {
      const int m  = mBase + mi * 16 + rHi + r;
      const int bb = m >> 11, ll = m & 2047;
      const float val = acc[t][r] + bias;
      if (sel == 0)
        qws[((size_t)(bb * H_ + hh) * L_ + ll) * DH_ + dh] = (_Float16)(val * 0.125f);
      else if (sel == 1)
        kws[((size_t)(bb * H_ + hh) * L_ + ll) * DH_ + dh] = (_Float16)val;
      else
        vtws[((size_t)(bb * H_ + hh) * DH_ + dh) * L_ + ll] = (_Float16)val;
    }
  }
}

// ---------------------------------------------------------------------------
// Kernel 2: flash attention.  One (b, h, 128 q-rows) per 256-thread block;
// each wave owns a 16-row Q tile.  K/V chunks staged to LDS by the TDM.
// ---------------------------------------------------------------------------
__global__ __launch_bounds__(256) void attn_wmma_kernel(
    const _Float16* __restrict__ qws, const _Float16* __restrict__ kws,
    const _Float16* __restrict__ vtws, const float* __restrict__ mask,
    float* __restrict__ out)
{
  __shared__ _Float16 Ks[32 * 64];     // K chunk [key][d]  (natural K layout)
  __shared__ _Float16 Vt[64 * 32];     // V chunk [dh][key] (natural V^T layout)
  __shared__ _Float16 Ps[8][16 * 32];  // per-wave P tile [m][key] (D->A relayout)

  const int tid  = threadIdx.x;
  const int lane = tid & 31;
  const int wave = tid >> 5;

  const int qBase = blockIdx.x * 128 + wave * 16;
  const int h = blockIdx.y;
  const int b = blockIdx.z;

  const size_t bhBase = (size_t)(b * H_ + h) * (size_t)(L_ * DH_);

  const int am  = lane & 15;           // A-frag row
  const int khi = (lane >> 4) * 8;     // A-frag k-halves base
  const int k16 = (lane >> 4) * 8;     // B-frag k base
  const int nl  = lane & 15;           // D/B column within subtile
  const int rHi = (lane >> 4) * 8;     // D row group

  // Q A-fragments (Dh = 64 = two 32-wide k-steps), resident for whole key loop
  const _Float16* qrow = qws + bhBase + (size_t)(qBase + am) * DH_;
  const v16h AQ0 = mkv16(*(const v8h*)(qrow + khi),      *(const v8h*)(qrow + 16 + khi));
  const v16h AQ1 = mkv16(*(const v8h*)(qrow + 32 + khi), *(const v8h*)(qrow + 48 + khi));

  float mrow[8], lrow[8];
#pragma unroll
  for (int r = 0; r < 8; ++r) { mrow[r] = -3.0e38f; lrow[r] = 0.0f; }
  v8f o0 = {}, o1 = {}, o2 = {}, o3 = {};

  const unsigned ldsK = (unsigned)(unsigned long long)(void*)&Ks[0];
  const unsigned ldsV = (unsigned)(unsigned long long)(void*)&Vt[0];
  const unsigned long long gK = (unsigned long long)(const void*)(kws + bhBase);
  const unsigned long long gV = (unsigned long long)(const void*)(vtws + bhBase);

  const size_t maskB = (size_t)b * L_ * L_;

  for (int kb = 0; kb < L_; kb += 32) {
    // ---- TDM: stage K chunk (rows=key, len=64) and V^T chunk (rows=dh, len=32)
    if (wave == 0) {
      // K: tile 64 x 32 rows, row stride 64 -> contiguous [key][d] in LDS
      tdm_load_2d(ldsK, gK + (unsigned long long)kb * DH_ * 2,
                  /*tensor_d0=*/DH_, /*tensor_d1=*/L_,
                  /*tile_d0=*/DH_, /*tile_d1=*/32, /*stride0=*/DH_);
      // V^T: tile 32 x 64 rows, row stride L_ -> [dh][key] in LDS
      tdm_load_2d(ldsV, gV + (unsigned long long)kb * 2,
                  /*tensor_d0=*/L_, /*tensor_d1=*/DH_,
                  /*tile_d0=*/32, /*tile_d1=*/DH_, /*stride0=*/L_);
      __builtin_amdgcn_s_wait_tensorcnt(0);
    }
    __syncthreads();

    // ---- S = (Q/8) * K^T : 2 key-subtiles x 2 d-steps = 4 WMMA ----
    v8f s0 = {}, s1 = {};
    {
      const int bK0 = (0 * 16 + nl) * 64;
      const int bK1 = (1 * 16 + nl) * 64;
      v16h Bf;
      Bf = mkv16(*(const v8h*)&Ks[bK0 + k16],      *(const v8h*)&Ks[bK0 + k16 + 16]);
      s0 = wmma16(AQ0, Bf, s0);
      Bf = mkv16(*(const v8h*)&Ks[bK0 + 32 + k16], *(const v8h*)&Ks[bK0 + 32 + k16 + 16]);
      s0 = wmma16(AQ1, Bf, s0);
      Bf = mkv16(*(const v8h*)&Ks[bK1 + k16],      *(const v8h*)&Ks[bK1 + k16 + 16]);
      s1 = wmma16(AQ0, Bf, s1);
      Bf = mkv16(*(const v8h*)&Ks[bK1 + 32 + k16], *(const v8h*)&Ks[bK1 + 32 + k16 + 16]);
      s1 = wmma16(AQ1, Bf, s1);
    }

    // ---- online softmax (row stats via 16-lane shfl_xor reductions) ----
#pragma unroll
    for (int r = 0; r < 8; ++r) {
      const int qm = qBase + rHi + r;
      float v0 = s0[r] + mask[maskB + (size_t)qm * L_ + kb + nl];
      float v1 = s1[r] + mask[maskB + (size_t)qm * L_ + kb + 16 + nl];
      float mx = fmaxf(v0, v1);
      mx = fmaxf(mx, __shfl_xor(mx, 1, 32));
      mx = fmaxf(mx, __shfl_xor(mx, 2, 32));
      mx = fmaxf(mx, __shfl_xor(mx, 4, 32));
      mx = fmaxf(mx, __shfl_xor(mx, 8, 32));
      const float mnew  = fmaxf(mrow[r], mx);
      const float alpha = __expf(mrow[r] - mnew);
      const float p0 = __expf(v0 - mnew);
      const float p1 = __expf(v1 - mnew);
      float rs = p0 + p1;
      rs += __shfl_xor(rs, 1, 32);
      rs += __shfl_xor(rs, 2, 32);
      rs += __shfl_xor(rs, 4, 32);
      rs += __shfl_xor(rs, 8, 32);
      lrow[r] = lrow[r] * alpha + rs;
      mrow[r] = mnew;
      o0[r] *= alpha; o1[r] *= alpha; o2[r] *= alpha; o3[r] *= alpha;
      Ps[wave][(rHi + r) * 32 + nl]      = (_Float16)p0;
      Ps[wave][(rHi + r) * 32 + 16 + nl] = (_Float16)p1;
    }
    // per-wave LDS RAW across lanes: drain DS counter before re-reading Ps
    asm volatile("s_wait_dscnt 0" ::: "memory");
    __builtin_amdgcn_wave_barrier();

    // ---- O += P * V : 4 dh-subtiles = 4 WMMA ----
    {
      const v16h AP = mkv16(*(const v8h*)&Ps[wave][am * 32 + khi],
                            *(const v8h*)&Ps[wave][am * 32 + 16 + khi]);
      v16h Bf;
      Bf = mkv16(*(const v8h*)&Vt[(0 * 16 + nl) * 32 + k16],
                 *(const v8h*)&Vt[(0 * 16 + nl) * 32 + k16 + 16]);
      o0 = wmma16(AP, Bf, o0);
      Bf = mkv16(*(const v8h*)&Vt[(1 * 16 + nl) * 32 + k16],
                 *(const v8h*)&Vt[(1 * 16 + nl) * 32 + k16 + 16]);
      o1 = wmma16(AP, Bf, o1);
      Bf = mkv16(*(const v8h*)&Vt[(2 * 16 + nl) * 32 + k16],
                 *(const v8h*)&Vt[(2 * 16 + nl) * 32 + k16 + 16]);
      o2 = wmma16(AP, Bf, o2);
      Bf = mkv16(*(const v8h*)&Vt[(3 * 16 + nl) * 32 + k16],
                 *(const v8h*)&Vt[(3 * 16 + nl) * 32 + k16 + 16]);
      o3 = wmma16(AP, Bf, o3);
    }
    __syncthreads();
  }

  // ---- epilogue: normalize and merge heads -> out[b, l, h*64 + dh] f32 ----
#pragma unroll
  for (int r = 0; r < 8; ++r) {
    const int qm = qBase + rHi + r;
    const float inv = 1.0f / lrow[r];
    float* op = out + ((size_t)b * L_ + qm) * DIM_ + h * DH_;
    op[ 0 + nl] = o0[r] * inv;
    op[16 + nl] = o1[r] * inv;
    op[32 + nl] = o2[r] * inv;
    op[48 + nl] = o3[r] * inv;
  }
}

// ---------------------------------------------------------------------------
extern "C" void kernel_launch(void* const* d_in, const int* in_sizes, int n_in,
                              void* d_out, int out_size, void* d_ws, size_t ws_size,
                              hipStream_t stream) {
  (void)in_sizes; (void)n_in; (void)out_size; (void)ws_size;
  const float* x  = (const float*)d_in[0];
  const float* mk = (const float*)d_in[1];
  const float* wq = (const float*)d_in[2];
  const float* bq = (const float*)d_in[3];
  const float* wk = (const float*)d_in[4];
  const float* bk = (const float*)d_in[5];
  const float* wv = (const float*)d_in[6];
  const float* bv = (const float*)d_in[7];
  float* out = (float*)d_out;

  // workspace: f16 Q [b,h,l,dh], K [b,h,l,dh], V^T [b,h,dh,l]  (3 x 16 MB)
  _Float16* wsp  = (_Float16*)d_ws;
  const size_t perT = (size_t)B_ * H_ * L_ * DH_;  // 8,388,608 halves
  _Float16* qws  = wsp;
  _Float16* kws  = wsp + perT;
  _Float16* vtws = wsp + 2 * perT;

  dim3 g1(3 * DIM_ / 128, (B_ * L_) / 64);         // 24 x 128 blocks
  hipLaunchKernelGGL(qkv_wmma_kernel, g1, dim3(256), 0, stream,
                     x, wq, bq, wk, bk, wv, bv, qws, kws, vtws);

  dim3 g2(L_ / 128, H_, B_);                       // 16 x 16 x 4 blocks
  hipLaunchKernelGGL(attn_wmma_kernel, g2, dim3(256), 0, stream,
                     qws, kws, vtws, mk, out);
}
